// DigitCaps_66202625901008
// MI455X (gfx1250) — matrix operand order repaired and checked by reading, not compiled, not run on hardware
//
#include <hip/hip_runtime.h>
#include <math.h>

// CapsNet DigitCaps dynamic routing for MI455X (gfx1250, wave32).
// u_hat recomputed from x/W each routing pass (inputs 367 MB < u_hat 524 MB; x fits L2).
// Per-route 64x32 @ 32x160 GEMM on v_wmma_f32_16x16x32_f16 (K=32 == IN_CH, N-tile == capsule).
// Staging via Tensor Data Mover (tensor_load_to_lds), double-buffered LDS, TENSORcnt sync.

#define NUM_ROUTES 12800
#define NUM_CAPS   10
#define IN_CH      32
#define OUT_CH     16
#define BATCH      64
#define CO         160   // NUM_CAPS * OUT_CH
#define RPB        32    // routes per block
#define THREADS    256   // 8 waves

typedef __attribute__((ext_vector_type(16))) _Float16     v16h;
typedef __attribute__((ext_vector_type(8)))  float        v8f;
typedef __attribute__((ext_vector_type(4)))  unsigned int u32x4;
typedef __attribute__((ext_vector_type(8)))  int          i32x8;
typedef __attribute__((ext_vector_type(4)))  int          i32x4;

#if __has_builtin(__builtin_amdgcn_tensor_load_to_lds)
#define HAVE_TDM 1
#else
#define HAVE_TDM 0
#endif

#if HAVE_TDM
__device__ __forceinline__ void tdm_load_raw(u32x4 g0, i32x8 g1) {
    i32x4 z4 = {};
#if __clang_major__ >= 23
    i32x8 z8 = {};
    __builtin_amdgcn_tensor_load_to_lds(g0, g1, z4, z4, z8, 0);
#else
    __builtin_amdgcn_tensor_load_to_lds(g0, g1, z4, z4, 0);
#endif
}

// 2D tile DMA: tile0 elements per row, tile1 rows, row stride stride0 (f32 elements).
// D# packing per CDNA5 ISA 8.3/8.4: count=1, type=2, data_size=4B, no pad/iterate/multicast.
__device__ __forceinline__ void tdm_issue_2d(unsigned int lds_off, const float* gptr,
                                             unsigned int tile0, unsigned int tile1,
                                             unsigned long long stride0) {
    unsigned long long ga = (unsigned long long)(uintptr_t)gptr;
    u32x4 g0;
    g0[0] = 1u;                                                 // count=1 (valid, user)
    g0[1] = lds_off;                                            // lds_addr [63:32]
    g0[2] = (unsigned int)ga;                                   // global_addr[31:0]
    g0[3] = (unsigned int)((ga >> 32) & 0x1FFFFFFull)           // global_addr[56:32]
          | (2u << 30);                                         // type=2 (image)
    i32x8 g1;
    g1[0] = (int)(2u << 16);                                    // data_size=4B
    g1[1] = (int)((tile0 & 0xFFFFu) << 16);                     // tensor_dim0[15:0]
    g1[2] = (int)((tile1 & 0xFFFFu) << 16);                     // dim0[31:16]=0 | tensor_dim1[15:0]
    g1[3] = (int)((tile0 & 0xFFFFu) << 16);                     // dim1[31:16]=0 | tile_dim0
    g1[4] = (int)(tile1 & 0xFFFFu);                             // tile_dim1 | tile_dim2=0
    g1[5] = (int)(unsigned int)(stride0 & 0xFFFFFFFFull);       // dim0_stride[31:0]
    g1[6] = (int)(unsigned int)((stride0 >> 32) & 0xFFFFull);   // dim0_stride[47:32] | dim1_stride lo=0
    g1[7] = 0;                                                  // dim1_stride hi
    tdm_load_raw(g0, g1);
}

__device__ __forceinline__ void stage_route_tdm(const float* x, const float* W, int r,
                                                float* xbuf, float* wbuf) {
    // x tile: 64 rows x 32 f32, row stride NUM_ROUTES*IN_CH
    tdm_issue_2d((unsigned int)(uintptr_t)xbuf, x + (size_t)r * IN_CH,
                 IN_CH, BATCH, (unsigned long long)NUM_ROUTES * IN_CH);
    // W tile: 5120 contiguous f32
    tdm_issue_2d((unsigned int)(uintptr_t)wbuf, W + (size_t)r * CO * IN_CH,
                 CO * IN_CH, 1, CO * IN_CH);
}
#else
// Fallback: cooperative copy (all threads), still double-buffered.
__device__ __forceinline__ void stage_route_copy(const float* __restrict__ x,
                                                 const float* __restrict__ W, int r,
                                                 float* xbuf, float* wbuf) {
    for (int idx = threadIdx.x; idx < BATCH * 8; idx += THREADS) {
        int b = idx >> 3, q = idx & 7;
        *(float4*)(xbuf + b * IN_CH + q * 4) =
            *(const float4*)(x + ((size_t)b * NUM_ROUTES + r) * IN_CH + q * 4);
    }
    const float4* wb = (const float4*)(W + (size_t)r * CO * IN_CH);
    for (int idx = threadIdx.x; idx < (CO * IN_CH) / 4; idx += THREADS)
        *(float4*)(wbuf + idx * 4) = wb[idx];
}
#endif

// A fragment (16x32 f16) from f32 LDS: lane<16 K{0..7,16..23}, lane>=16 K{8..15,24..31}
__device__ __forceinline__ v16h load_A32(const float* xs, int mtile, int lane) {
    int row = mtile * 16 + (lane & 15);
    int kb  = (lane >= 16) ? 8 : 0;
    const float* p = xs + row * IN_CH + kb;
    float4 f0 = *(const float4*)(p);
    float4 f1 = *(const float4*)(p + 4);
    float4 f2 = *(const float4*)(p + 16);
    float4 f3 = *(const float4*)(p + 20);
    v16h a;
    a[0]=(_Float16)f0.x;  a[1]=(_Float16)f0.y;  a[2]=(_Float16)f0.z;  a[3]=(_Float16)f0.w;
    a[4]=(_Float16)f1.x;  a[5]=(_Float16)f1.y;  a[6]=(_Float16)f1.z;  a[7]=(_Float16)f1.w;
    a[8]=(_Float16)f2.x;  a[9]=(_Float16)f2.y;  a[10]=(_Float16)f2.z; a[11]=(_Float16)f2.w;
    a[12]=(_Float16)f3.x; a[13]=(_Float16)f3.y; a[14]=(_Float16)f3.z; a[15]=(_Float16)f3.w;
    return a;
}

// B fragment (32x16 f16): N = lane&15, lanes 0-15 K{0..7,16..23}, lanes 16-31 K{8..15,24..31}
__device__ __forceinline__ v16h load_B32(const float* wl, int cap, int lane) {
    int col = cap * 16 + (lane & 15);
    int kb  = (lane >= 16) ? 8 : 0;
    const float* p = wl + col * IN_CH + kb;
    float4 f0 = *(const float4*)(p);
    float4 f1 = *(const float4*)(p + 4);
    float4 f2 = *(const float4*)(p + 16);
    float4 f3 = *(const float4*)(p + 20);
    v16h b;
    b[0]=(_Float16)f0.x;  b[1]=(_Float16)f0.y;  b[2]=(_Float16)f0.z;  b[3]=(_Float16)f0.w;
    b[4]=(_Float16)f1.x;  b[5]=(_Float16)f1.y;  b[6]=(_Float16)f1.z;  b[7]=(_Float16)f1.w;
    b[8]=(_Float16)f2.x;  b[9]=(_Float16)f2.y;  b[10]=(_Float16)f2.z; b[11]=(_Float16)f2.w;
    b[12]=(_Float16)f3.x; b[13]=(_Float16)f3.y; b[14]=(_Float16)f3.z; b[15]=(_Float16)f3.w;
    return b;
}

// ---- s[b,c,o] += sum_r coef[r,c] * u_hat[b,r,c,o] ---------------------------
template <bool HAS_COEF>
__global__ __launch_bounds__(THREADS)
void caps_sum_kernel(const float* __restrict__ x, const float* __restrict__ W,
                     const float* __restrict__ coef, float* __restrict__ s_out) {
    __shared__ float xbuf[2][BATCH * IN_CH];   // 2 x 8 KB
    __shared__ float wbuf[2][CO * IN_CH];      // 2 x 20 KB
    const int lane  = threadIdx.x & 31;
    const int wave  = threadIdx.x >> 5;
    const int mtile = wave & 3;
    const int cbase = (wave >> 2) * 5;
    const int r0    = blockIdx.x * RPB;
    int cur = 0;

#if HAVE_TDM
    if (wave == 0) stage_route_tdm(x, W, r0, xbuf[0], wbuf[0]);
#else
    stage_route_copy(x, W, r0, xbuf[0], wbuf[0]);
#endif

    v8f acc[5] = {};
    for (int rr = 0; rr < RPB; ++rr) {
#if HAVE_TDM
        if (wave == 0) __builtin_amdgcn_s_wait_tensorcnt(0);
#endif
        __syncthreads();   // buffer `cur` ready for everyone; prior reads of cur^1 done
        if (rr + 1 < RPB) {
#if HAVE_TDM
            if (wave == 0) stage_route_tdm(x, W, r0 + rr + 1, xbuf[cur ^ 1], wbuf[cur ^ 1]);
#else
            stage_route_copy(x, W, r0 + rr + 1, xbuf[cur ^ 1], wbuf[cur ^ 1]);
#endif
        }
        float cw[5];
        if (HAS_COEF) {
            #pragma unroll
            for (int ci = 0; ci < 5; ++ci)
                cw[ci] = coef[(r0 + rr) * NUM_CAPS + cbase + ci];
        }
        v16h a = load_A32(xbuf[cur], mtile, lane);
        #pragma unroll
        for (int ci = 0; ci < 5; ++ci) {
            v16h b = load_B32(wbuf[cur], cbase + ci, lane);
            v8f zero = {};
            v8f t = __builtin_amdgcn_wmma_f32_16x16x32_f16(
                        false, a, false, b, (short)0, zero, false, false);
            #pragma unroll
            for (int v = 0; v < 8; ++v) {
                float wgt = HAS_COEF ? cw[ci] : 1.0f;
                acc[ci][v] = fmaf(wgt, t[v], acc[ci][v]);
            }
        }
        cur ^= 1;
    }
    const int bhi = (lane >= 16) ? 8 : 0;
    #pragma unroll
    for (int ci = 0; ci < 5; ++ci) {
        int co = (cbase + ci) * 16 + (lane & 15);
        #pragma unroll
        for (int v = 0; v < 8; ++v) {
            int b = mtile * 16 + v + bhi;
            __hip_atomic_fetch_add(&s_out[b * CO + co], acc[ci][v],
                                   __ATOMIC_RELAXED, __HIP_MEMORY_SCOPE_AGENT);
        }
    }
}

// ---- a[r,c] = sum_{b,o} u_hat[b,r,c,o] * v[b,c,o] ---------------------------
__global__ __launch_bounds__(THREADS)
void caps_agree_kernel(const float* __restrict__ x, const float* __restrict__ W,
                       const float* __restrict__ v, int accumulate,
                       float* __restrict__ b_arr) {
    __shared__ float xbuf[2][BATCH * IN_CH];
    __shared__ float wbuf[2][CO * IN_CH];
    __shared__ float a_lds[NUM_CAPS];
    const int lane  = threadIdx.x & 31;
    const int wave  = threadIdx.x >> 5;
    const int mtile = wave & 3;
    const int cbase = (wave >> 2) * 5;
    const int r0    = blockIdx.x * RPB;
    const int bhi   = (lane >= 16) ? 8 : 0;
    int cur = 0;

    // v values this lane ever needs: kept in registers for the whole kernel.
    float vreg[5][8];
    #pragma unroll
    for (int ci = 0; ci < 5; ++ci)
        #pragma unroll
        for (int vv = 0; vv < 8; ++vv)
            vreg[ci][vv] = v[(mtile * 16 + vv + bhi) * CO + (cbase + ci) * 16 + (lane & 15)];

    if (threadIdx.x < NUM_CAPS) a_lds[threadIdx.x] = 0.f;
#if HAVE_TDM
    if (wave == 0) stage_route_tdm(x, W, r0, xbuf[0], wbuf[0]);
#else
    stage_route_copy(x, W, r0, xbuf[0], wbuf[0]);
#endif

    for (int rr = 0; rr < RPB; ++rr) {
        int r = r0 + rr;
#if HAVE_TDM
        if (wave == 0) __builtin_amdgcn_s_wait_tensorcnt(0);
#endif
        __syncthreads();
        if (rr + 1 < RPB) {
#if HAVE_TDM
            if (wave == 0) stage_route_tdm(x, W, r + 1, xbuf[cur ^ 1], wbuf[cur ^ 1]);
#else
            stage_route_copy(x, W, r + 1, xbuf[cur ^ 1], wbuf[cur ^ 1]);
#endif
        }
        v16h a = load_A32(xbuf[cur], mtile, lane);
        // 1) all 5 capsule partials first
        float p[5];
        #pragma unroll
        for (int ci = 0; ci < 5; ++ci) {
            v16h b = load_B32(wbuf[cur], cbase + ci, lane);
            v8f zero = {};
            v8f t = __builtin_amdgcn_wmma_f32_16x16x32_f16(
                        false, a, false, b, (short)0, zero, false, false);
            float acc = t[0] * vreg[ci][0];
            #pragma unroll
            for (int vv = 1; vv < 8; ++vv) acc = fmaf(t[vv], vreg[ci][vv], acc);
            p[ci] = acc;
        }
        // 2) butterfly reduce, interleaved across capsules so the 5 ds_bpermute
        //    per stage are independent and share one s_wait_dscnt.
        #pragma unroll
        for (int off = 16; off > 0; off >>= 1) {
            float q[5];
            #pragma unroll
            for (int ci = 0; ci < 5; ++ci) q[ci] = __shfl_xor(p[ci], off, 32);
            #pragma unroll
            for (int ci = 0; ci < 5; ++ci) p[ci] += q[ci];
        }
        if (lane == 0) {
            #pragma unroll
            for (int ci = 0; ci < 5; ++ci)
                atomicAdd(&a_lds[cbase + ci], p[ci]);   // ds_add_f32
        }
        __syncthreads();                   // all ds atomics complete
        if (threadIdx.x < NUM_CAPS) {      // same thread reads & rezeros: no race
            float val = a_lds[threadIdx.x];
            float* dst = &b_arr[(size_t)r * NUM_CAPS + threadIdx.x];
            *dst = accumulate ? (*dst + val) : val;
            a_lds[threadIdx.x] = 0.f;
        }
        cur ^= 1;
    }
}

// ---- softmax over routes (axis=0), one block per capsule --------------------
__global__ __launch_bounds__(256)
void softmax_routes_kernel(const float* __restrict__ b_arr, float* __restrict__ c_arr) {
    const int cap = blockIdx.x;
    const int tid = threadIdx.x;
    __shared__ float red[256];
    float m = -3.4e38f;
    for (int r = tid; r < NUM_ROUTES; r += 256)
        m = fmaxf(m, b_arr[r * NUM_CAPS + cap]);
    red[tid] = m; __syncthreads();
    for (int off = 128; off > 0; off >>= 1) {
        if (tid < off) red[tid] = fmaxf(red[tid], red[tid + off]);
        __syncthreads();
    }
    m = red[0]; __syncthreads();
    float s = 0.f;
    for (int r = tid; r < NUM_ROUTES; r += 256)
        s += expf(b_arr[r * NUM_CAPS + cap] - m);
    red[tid] = s; __syncthreads();
    for (int off = 128; off > 0; off >>= 1) {
        if (tid < off) red[tid] += red[tid + off];
        __syncthreads();
    }
    float inv = 1.f / red[0];
    for (int r = tid; r < NUM_ROUTES; r += 256)
        c_arr[r * NUM_CAPS + cap] = expf(b_arr[r * NUM_CAPS + cap] - m) * inv;
}

// ---- squash -----------------------------------------------------------------
__global__ void squash_kernel(const float* __restrict__ s_in,
                              float* __restrict__ v_out, float scale) {
    int i = blockIdx.x * blockDim.x + threadIdx.x;   // (b,c) pair
    if (i >= BATCH * NUM_CAPS) return;
    const float* s = s_in + i * OUT_CH;
    float tmp[OUT_CH]; float sq = 0.f;
    #pragma unroll
    for (int o = 0; o < OUT_CH; ++o) { float val = s[o] * scale; tmp[o] = val; sq = fmaf(val, val, sq); }
    float f = sq / ((1.f + sq) * sqrtf(sq));
    #pragma unroll
    for (int o = 0; o < OUT_CH; ++o) v_out[i * OUT_CH + o] = tmp[o] * f;
}

extern "C" void kernel_launch(void* const* d_in, const int* in_sizes, int n_in,
                              void* d_out, int out_size, void* d_ws, size_t ws_size,
                              hipStream_t stream) {
    const float* x = (const float*)d_in[0];   // [64, 12800, 32]
    const float* W = (const float*)d_in[1];   // [12800, 10, 16, 32]
    float* out = (float*)d_out;               // [64, 10, 16]

    float* s_acc = (float*)d_ws;                              // 10240 f
    float* v_arr = s_acc + BATCH * CO;                        // 10240 f
    float* b_arr = v_arr + BATCH * CO;                        // 128000 f
    float* c_arr = b_arr + (size_t)NUM_ROUTES * NUM_CAPS;     // 128000 f

    dim3 hgrid(NUM_ROUTES / RPB), hblk(THREADS);
    dim3 sgrid((BATCH * NUM_CAPS + 255) / 256), sblk(256);

    // iter 0: c uniform (1/R folded into squash scale)
    hipMemsetAsync(s_acc, 0, BATCH * CO * sizeof(float), stream);
    caps_sum_kernel<false><<<hgrid, hblk, 0, stream>>>(x, W, nullptr, s_acc);
    squash_kernel<<<sgrid, sblk, 0, stream>>>(s_acc, v_arr, 1.0f / NUM_ROUTES);
    caps_agree_kernel<<<hgrid, hblk, 0, stream>>>(x, W, v_arr, 0, b_arr);
    softmax_routes_kernel<<<dim3(NUM_CAPS), dim3(256), 0, stream>>>(b_arr, c_arr);

    // iter 1
    hipMemsetAsync(s_acc, 0, BATCH * CO * sizeof(float), stream);
    caps_sum_kernel<true><<<hgrid, hblk, 0, stream>>>(x, W, c_arr, s_acc);
    squash_kernel<<<sgrid, sblk, 0, stream>>>(s_acc, v_arr, 1.0f);
    caps_agree_kernel<<<hgrid, hblk, 0, stream>>>(x, W, v_arr, 1, b_arr);
    softmax_routes_kernel<<<dim3(NUM_CAPS), dim3(256), 0, stream>>>(b_arr, c_arr);

    // iter 2: final v -> d_out
    hipMemsetAsync(s_acc, 0, BATCH * CO * sizeof(float), stream);
    caps_sum_kernel<true><<<hgrid, hblk, 0, stream>>>(x, W, c_arr, s_acc);
    squash_kernel<<<sgrid, sblk, 0, stream>>>(s_acc, out, 1.0f);
}
